// Model_50337016709121
// MI455X (gfx1250) — matrix operand order repaired and checked by reading, not compiled, not run on hardware
//
#include <hip/hip_runtime.h>
#include <hip/hip_bf16.h>

// Feature-wise fused MLP for MI455X (gfx1250, wave32, WMMA bf16).
// h = relu(x*w1+b1); h = relu(h@W2+b2); h = relu(h@W3+b3); xp = h.w4+b4;
// out = sum_f xp[:,f]*lr_w[f] + lr_b
//
// One workgroup per (batch-chunk, feature). Weights converted to bf16,
// transposed into LDS once, reused across 8 batch tiles of 128 rows.
// Layer-1 A-fragments are synthesized in registers (rank-1), so the tile
// loop is barrier-free: each wave only touches its own 16 rows of H.

typedef __bf16 bf16_t;
typedef __attribute__((ext_vector_type(16))) __bf16 v16bf;
typedef __attribute__((ext_vector_type(8)))  __bf16 v8bf;
typedef __attribute__((ext_vector_type(8)))  float   v8f;

#define FDIM   256
#define NDIM   128
#define BDIM   4096
#define MTILE  128
#define PITCH  136          // bf16 elems: 272B row stride -> every row 16B aligned
#define CHUNKS 4
#define TILES_PER_WG ((BDIM / MTILE) / CHUNKS)

__device__ __forceinline__ v16bf frag_cat(v8bf lo, v8bf hi) {
  return __builtin_shufflevector(lo, hi, 0,1,2,3,4,5,6,7,8,9,10,11,12,13,14,15);
}

// A fragment (16-bit A 16x32 layout, ISA 7.12.2):
// lane ln holds row M; sel = lane>>4.
// elems 0..7  = K[kb + sel*8      .. +8)
// elems 8..15 = K[kb + 16 + sel*8 .. +8)
__device__ __forceinline__ v16bf load_fragA(const bf16_t* rowp, int kb, int sel) {
  const bf16_t* p = rowp + kb + sel * 8;
  v8bf lo = *(const v8bf*)(p);
  v8bf hi = *(const v8bf*)(p + 16);
  return frag_cat(lo, hi);
}

// B fragment (16-bit B 32x16): lane holds column N = ln; same K order as A:
// elems 0..7 = K[kb + sel*8 .. +8), elems 8..15 = K[kb + 16 + sel*8 .. +8).
__device__ __forceinline__ v16bf load_fragB(const bf16_t* rowp, int kb, int sel) {
  const bf16_t* p = rowp + kb + sel * 8;
  v8bf lo = *(const v8bf*)(p);
  v8bf hi = *(const v8bf*)(p + 16);
  return frag_cat(lo, hi);
}

__global__ void init_out_kernel(float* __restrict__ out,
                                const float* __restrict__ lr_b, int n) {
  int i = blockIdx.x * blockDim.x + threadIdx.x;
  if (i < n) out[i] = lr_b[0];
}

__global__ __launch_bounds__(256)
void fmlp_wmma_kernel(const float* __restrict__ x,
                      const float* __restrict__ w1, const float* __restrict__ b1,
                      const float* __restrict__ w2, const float* __restrict__ b2,
                      const float* __restrict__ w3, const float* __restrict__ b3,
                      const float* __restrict__ w4, const float* __restrict__ b4,
                      const float* __restrict__ lr_w,
                      float* __restrict__ out)
{
  extern __shared__ char smem[];
  bf16_t* Wt2 = (bf16_t*)smem;               // [128][PITCH] : W2^T (o-major)
  bf16_t* Wt3 = Wt2 + NDIM * PITCH;          // [128][PITCH] : W3^T
  bf16_t* Hb  = Wt3 + NDIM * PITCH;          // [128][PITCH] : activations
  float*  w1s = (float*)(Hb + MTILE * PITCH);  // 32B-aligned (offset 104448)
  float*  b1s = w1s + NDIM;
  float*  b2s = b1s + NDIM;
  float*  b3s = b2s + NDIM;
  float*  w4s = b3s + NDIM;

  const int t    = threadIdx.x;
  const int wav  = t >> 5;
  const int lane = t & 31;
  const int sel  = lane >> 4;   // which 16-lane half of the wave
  const int ln   = lane & 15;
  const int f    = blockIdx.y;

  // ---------------- per-feature setup (once) ----------------
  if (t < NDIM) {
    w1s[t] = w1[f * NDIM + t];
    b1s[t] = b1[f * NDIM + t];
    b2s[t] = b2[f * NDIM + t];
    b3s[t] = b3[f * NDIM + t];
    w4s[t] = w4[f * NDIM + t];
  }
  // Transpose + convert W2[f], W3[f] to bf16 in LDS: Wt[o][i] = W[i][o].
  // Coalesced global reads; strided LDS writes (one-time cost per WG).
  const float* w2f = w2 + (size_t)f * NDIM * NDIM;
  const float* w3f = w3 + (size_t)f * NDIM * NDIM;
  #pragma unroll 4
  for (int e = 0; e < (NDIM * NDIM) / 256; ++e) {
    int idx = e * 256 + t;
    int i = idx >> 7, o = idx & 127;
    Wt2[o * PITCH + i] = (bf16_t)w2f[idx];
    Wt3[o * PITCH + i] = (bf16_t)w3f[idx];
  }
  __syncthreads();   // only barrier in the kernel

  const float b4f = b4[f];
  const float lwf = lr_w[f];
  const int   row0 = wav * 16;                    // this wave's 16-row strip
  const bf16_t* Arow = Hb + (row0 + ln) * PITCH;  // lane's A-matrix row
  const v8f vzero = {0.f, 0.f, 0.f, 0.f, 0.f, 0.f, 0.f, 0.f};

  for (int tile = 0; tile < TILES_PER_WG; ++tile) {
    const int m0 = (blockIdx.x * TILES_PER_WG + tile) * MTILE;

    // Per-lane x scalar for this lane's A-matrix row (row = row0 + ln).
    const float xr = x[(size_t)(m0 + row0 + ln) * FDIM + f];

    // ---- layer 2: acc = relu(x*w1+b1) @ W2, A synthesized in registers ----
    v8f acc[8];
    #pragma unroll
    for (int n = 0; n < 8; ++n) acc[n] = vzero;
    #pragma unroll
    for (int kb = 0; kb < NDIM; kb += 32) {
      // A-fragment K runs: [kb+sel*8, +8) and [kb+16+sel*8, +8)
      const v8f wlo = *(const v8f*)(w1s + kb + sel * 8);
      const v8f blo = *(const v8f*)(b1s + kb + sel * 8);
      const v8f whi = *(const v8f*)(w1s + kb + 16 + sel * 8);
      const v8f bhi = *(const v8f*)(b1s + kb + 16 + sel * 8);
      v8f hlo, hhi;
      #pragma unroll
      for (int j = 0; j < 8; ++j) {
        hlo[j] = fmaxf(xr * wlo[j] + blo[j], 0.f);
        hhi[j] = fmaxf(xr * whi[j] + bhi[j], 0.f);
      }
      v16bf a = frag_cat(__builtin_convertvector(hlo, v8bf),
                         __builtin_convertvector(hhi, v8bf));
      #pragma unroll
      for (int n = 0; n < 8; ++n) {
        v16bf b = load_fragB(Wt2 + (n * 16 + ln) * PITCH, kb, sel);
        acc[n] = __builtin_amdgcn_wmma_f32_16x16x32_bf16(
            false, a, false, b, (short)0, acc[n], false, false);
      }
    }
    // C layout: vgpr g -> row g + 8*sel, col = n*16 + ln. Rows are
    // wave-private, so no barrier needed (same-wave DS is in-order).
    #pragma unroll
    for (int n = 0; n < 8; ++n) {
      float bias = b2s[n * 16 + ln];
      #pragma unroll
      for (int g = 0; g < 8; ++g) {
        float v = fmaxf(acc[n][g] + bias, 0.f);
        Hb[(row0 + g + 8 * sel) * PITCH + n * 16 + ln] = (bf16_t)v;
      }
    }

    // ---- layer 3: acc = H @ W3 (kept in registers) ----
    #pragma unroll
    for (int n = 0; n < 8; ++n) acc[n] = vzero;
    #pragma unroll
    for (int kb = 0; kb < NDIM; kb += 32) {
      v16bf a = load_fragA(Arow, kb, sel);
      #pragma unroll
      for (int n = 0; n < 8; ++n) {
        v16bf b = load_fragB(Wt3 + (n * 16 + ln) * PITCH, kb, sel);
        acc[n] = __builtin_amdgcn_wmma_f32_16x16x32_bf16(
            false, a, false, b, (short)0, acc[n], false, false);
      }
    }

    // ---- layer 4 + head: xp = relu(acc+b3) . w4 + b4 ; out += xp*lr_w[f] ----
    float p[8];
    #pragma unroll
    for (int g = 0; g < 8; ++g) p[g] = 0.f;
    #pragma unroll
    for (int n = 0; n < 8; ++n) {
      float bias = b3s[n * 16 + ln];
      float wv   = w4s[n * 16 + ln];
      #pragma unroll
      for (int g = 0; g < 8; ++g)
        p[g] += fmaxf(acc[n][g] + bias, 0.f) * wv;
    }
    #pragma unroll
    for (int g = 0; g < 8; ++g) {
      float v = p[g];
      v += __shfl_xor(v, 1);
      v += __shfl_xor(v, 2);
      v += __shfl_xor(v, 4);
      v += __shfl_xor(v, 8);     // full sum within each 16-lane half
      if (ln == 0) {
        int row = row0 + g + 8 * sel;
        atomicAdd(&out[m0 + row], (v + b4f) * lwf);
      }
    }
  }
}

extern "C" void kernel_launch(void* const* d_in, const int* in_sizes, int n_in,
                              void* d_out, int out_size, void* d_ws, size_t ws_size,
                              hipStream_t stream) {
  const float* x    = (const float*)d_in[0];
  const float* w1   = (const float*)d_in[1];
  const float* b1   = (const float*)d_in[2];
  const float* w2   = (const float*)d_in[3];
  const float* b2   = (const float*)d_in[4];
  const float* w3   = (const float*)d_in[5];
  const float* b3   = (const float*)d_in[6];
  const float* w4   = (const float*)d_in[7];
  const float* b4   = (const float*)d_in[8];
  const float* lr_w = (const float*)d_in[9];
  const float* lr_b = (const float*)d_in[10];
  float* out = (float*)d_out;

  // out is poisoned by the harness: seed with lr_b, then accumulate.
  init_out_kernel<<<(BDIM + 255) / 256, 256, 0, stream>>>(out, lr_b, BDIM);

  size_t shmem = (size_t)(2 * NDIM * PITCH + MTILE * PITCH) * sizeof(bf16_t)
               + (size_t)(5 * NDIM) * sizeof(float);   // ~104.5 KB
  dim3 grid(CHUNKS, FDIM);
  fmlp_wmma_kernel<<<grid, 256, shmem, stream>>>(
      x, w1, b1, w2, b2, w3, b3, w4, b4, lr_w, out);
}
